// TimeSformer_with_AA_46591805227447
// MI455X (gfx1250) — compile-verified, matrix operand order checked by reading, not run on hardware
//
#include <hip/hip_runtime.h>
#include <cstdint>
#include <cstddef>

typedef __attribute__((ext_vector_type(16))) _Float16 v16h;
typedef __attribute__((ext_vector_type(8)))  _Float16 v8h;
typedef __attribute__((ext_vector_type(4)))  _Float16 v4h;
typedef __attribute__((ext_vector_type(8)))  float    v8f;
typedef __attribute__((ext_vector_type(4)))  float    v4f;
typedef __attribute__((ext_vector_type(4)))  unsigned int u32x4;
typedef __attribute__((ext_vector_type(8)))  int      i32x8;
typedef __attribute__((ext_vector_type(4)))  int      i32x4;

union HF16 { v16h v; v8h h[2]; };

#if defined(__has_builtin)
#if __has_builtin(__builtin_amdgcn_tensor_load_to_lds) && __has_builtin(__builtin_amdgcn_s_wait_tensorcnt)
#define USE_TDM 1
#endif
#endif
#ifndef USE_TDM
#define USE_TDM 0
#endif

#define SEQ    1570
#define NTOK   1568
#define NB     4
#define NH     8
#define NBH    32
#define DHEAD  64
#define DMODEL 512
#define NROWS  (NB*SEQ)     /* 6280 */
#define NPATCH 196
#define NF     8

// ---------------------------------------------------------------------------
// WMMA GEMM: C[M,N](f32) = A[M,K](f16) * B[K,N](f16) (+bias) (+resid).
// Block tile 256x64, K-step 32, 8 waves (wave32); each wave owns two 16-row
// strips x 64 cols (8 WMMA accumulators). Requires K % 32 == 0.
// Double-buffered LDS: A tile DMA'd by the Tensor Data Mover (TENSORcnt
// pipelined against compute) with LDS row padding 32+8 halves; B tile staged
// transposed for the K-major-per-lane fragment layout.
// ---------------------------------------------------------------------------
#define BM 256
#define BN 64
#define BK 32
#define LSTR 40            // f16 per LDS row (32 + 8 pad) == 16 DW + 4 DW pad
#define ATILE (BM*LSTR)    // 10240 halves per buffer
#define BTILE (BN*LSTR)    // 2560 halves per buffer

__global__ __launch_bounds__(256) void k_gemm(const _Float16* __restrict__ A,
    const _Float16* __restrict__ B, const float* __restrict__ bias,
    const float* resid, float* C,
    int M, int N, int K, int flags)
{
  __shared__ _Float16 As[2*ATILE];
  __shared__ _Float16 Bs[2*BTILE];
  const int tid  = threadIdx.x;
  const int wv   = tid >> 5;
  const int lane = tid & 31;
  const int g    = lane >> 4;   // lane group (K-half select per ISA layout)
  const int r    = lane & 15;   // row (A) / col (B) within 16x16 tile
  const int m0   = blockIdx.y * BM;
  const int n0   = blockIdx.x * BN;

  v8f acc[2][4];
#pragma unroll
  for (int st = 0; st < 2; st++)
#pragma unroll
    for (int j = 0; j < 4; j++)
#pragma unroll
      for (int e = 0; e < 8; e++) acc[st][j][e] = 0.f;

#if USE_TDM
  auto tdmA = [&](int k0, int aofs) {
    // Tensor Data Mover: DMA A tile (<=256 x 32 f16) into padded LDS rows
    unsigned long long ga = (unsigned long long)(uintptr_t)(A + (size_t)m0*K + k0);
    unsigned int ldsoff   = (unsigned int)(uintptr_t)(void*)&As[aofs];
    int rows = M - m0; if (rows > BM) rows = BM;     // OOB rows -> zero-fill
    unsigned int td0 = (unsigned int)(K - k0);
    unsigned long long str = (unsigned long long)K;  // dim0 stride (elements)
    u32x4 g0;
    g0[0] = 1u;                                      // count=1, user mode
    g0[1] = ldsoff;                                  // lds_addr (bytes)
    g0[2] = (unsigned int)ga;                        // global_addr lo
    g0[3] = (unsigned int)((ga >> 32) & 0x1FFFFFFull) | (2u << 30); // hi|type=2
    i32x8 g1;
    g1[0] = (int)((1u<<16)        // data_size = 1 -> 2 bytes
                | (1u<<20)        // pad_enable
                | (3u<<22)        // pad_interval: 16 DWORDs
                | (3u<<25));      // pad_amount: 4 DWORDs
    g1[1] = (int)((td0 & 0xFFFFu) << 16);                              // dim0 lo16
    g1[2] = (int)(((td0 >> 16) & 0xFFFFu) | (((unsigned)rows) << 16)); // dim0 hi | dim1 lo
    g1[3] = (int)((32u << 16));                                        // dim1 hi=0 | tile_dim0=32
    g1[4] = (int)((unsigned)BM);                                       // tile_dim1=256, tile_dim2=0
    g1[5] = (int)(unsigned int)(str & 0xFFFFFFFFull);                  // dim0_stride lo
    g1[6] = (int)(unsigned int)((str >> 32) & 0xFFFFull);              // dim0_stride hi
    g1[7] = 0;
    i32x4 z4; z4[0] = z4[1] = z4[2] = z4[3] = 0;
#if defined(__clang_major__) && (__clang_major__ >= 23)
    i32x8 z8;
#pragma unroll
    for (int e = 0; e < 8; e++) z8[e] = 0;
    __builtin_amdgcn_tensor_load_to_lds(g0, g1, z4, z4, z8, 0);
#else
    __builtin_amdgcn_tensor_load_to_lds(g0, g1, z4, z4, 0);
#endif
  };
#else
  auto stageA = [&](int k0, int aofs) {   // 1024 x (8-half) vectors, 4/thread
    for (int i = tid; i < (BM*BK)/8; i += 256) {
      int row = i >> 2, c8 = (i & 3)*8;
      int gm = m0 + row;
      v8h v;
#pragma unroll
      for (int e = 0; e < 8; e++) v[e] = (_Float16)0.f;
      if (gm < M) v = *(const v8h*)(A + (size_t)gm*K + k0 + c8);
      *(v8h*)(&As[aofs + row*LSTR + c8]) = v;
    }
  };
#endif
  auto stageB = [&](int k0, int bofs) {   // 256 x (8-half) vectors, 1/thread
    for (int i = tid; i < (BK*BN)/8; i += 256) {
      int kk = i >> 3, n8 = (i & 7)*8;
      int gn = n0 + n8;
      v8h v;
#pragma unroll
      for (int e = 0; e < 8; e++) v[e] = (_Float16)0.f;
      if (gn < N) v = *(const v8h*)(B + (size_t)(k0+kk)*N + gn);
#pragma unroll
      for (int e = 0; e < 8; e++) Bs[bofs + (n8+e)*LSTR + kk] = v[e];
    }
  };
  auto compute = [&](int aofs, int bofs) {
    // A fragments: lane r row, K = g*8..g*8+7 then 16+g*8..16+g*8+7
    HF16 af[2];
#pragma unroll
    for (int st = 0; st < 2; st++) {
      const _Float16* ap = &As[aofs + (st*128 + wv*16 + r)*LSTR + g*8];
      af[st].h[0] = *(const v8h*)(ap);
      af[st].h[1] = *(const v8h*)(ap + 16);
    }
#pragma unroll
    for (int j = 0; j < 4; j++) {
      HF16 bf;
      const _Float16* bp = &Bs[bofs + (j*16 + r)*LSTR + g*8];
      bf.h[0] = *(const v8h*)(bp);
      bf.h[1] = *(const v8h*)(bp + 16);
#pragma unroll
      for (int st = 0; st < 2; st++)
        acc[st][j] = __builtin_amdgcn_wmma_f32_16x16x32_f16(
            false, af[st].v, false, bf.v, (short)0, acc[st][j], false, false);
    }
  };

  const int T = K / BK;
  // ---- prologue: fill buffer 0
#if USE_TDM
  if (wv == 0) tdmA(0, 0);
#else
  stageA(0, 0);
#endif
  stageB(0, 0);
#if USE_TDM
  if (wv == 0) __builtin_amdgcn_s_wait_tensorcnt(0);
#endif
  __syncthreads();

  int c = 0;
  for (int t = 0; t < T; t++) {
    int nc = c ^ 1;
    if (t + 1 < T) {
      // producer for tile t+1 runs concurrently with compute of tile t
#if USE_TDM
      if (wv == 0) tdmA((t+1)*BK, nc*ATILE);
#else
      stageA((t+1)*BK, nc*ATILE);
#endif
      stageB((t+1)*BK, nc*BTILE);
    }
    compute(c*ATILE, c*BTILE);
#if USE_TDM
    if (wv == 0) __builtin_amdgcn_s_wait_tensorcnt(0);  // t+1's DMA, post-compute
#endif
    __syncthreads();
    c = nc;
  }

  // epilogue: D layout — VGPR e holds row e + 8*g, col = lane%16
#pragma unroll
  for (int st = 0; st < 2; st++) {
#pragma unroll
    for (int j = 0; j < 4; j++) {
      int col = n0 + j*16 + r;
      if (col < N) {
        float bv = (flags & 1) ? bias[col] : 0.f;
#pragma unroll
        for (int e = 0; e < 8; e++) {
          int row = m0 + st*128 + wv*16 + e + 8*g;
          if (row < M) {
            float v = acc[st][j][e] + bv;
            if (flags & 2) v += resid[(size_t)row*N + col];
            C[(size_t)row*N + col] = v;
          }
        }
      }
    }
  }
}

// ---------------------------------------------------------------------------
// f32 -> f16 cast, 4 elements per thread (for weight matrices)
// ---------------------------------------------------------------------------
__global__ void k_cast_f16(const float* __restrict__ in, _Float16* __restrict__ out, int n4)
{
  int idx = blockIdx.x*256 + threadIdx.x;
  if (idx >= n4) return;
  v4f f = *(const v4f*)(in + (size_t)idx*4);
  v4h h;
#pragma unroll
  for (int e = 0; e < 4; e++) h[e] = (_Float16)f[e];
  *(v4h*)(out + (size_t)idx*4) = h;
}

// ---------------------------------------------------------------------------
// LayerNorm: one wave32 per 512-wide row; OT = f16 (GEMM feed) or f32 (head)
// ---------------------------------------------------------------------------
template <typename OT>
__global__ __launch_bounds__(256) void k_layernorm(const float* __restrict__ in,
    long long instride, const float* __restrict__ gam, const float* __restrict__ bet,
    OT* __restrict__ out, long long outstride, int rows)
{
  int wv = threadIdx.x >> 5, lane = threadIdx.x & 31;
  int row = blockIdx.x*8 + wv;
  if (row >= rows) return;
  const float* p = in + (size_t)row*instride;
  float s = 0.f, ss = 0.f;
#pragma unroll
  for (int i = 0; i < 16; i++) { float v = p[lane + i*32]; s += v; ss += v*v; }
#pragma unroll
  for (int m = 16; m > 0; m >>= 1) { s += __shfl_xor(s, m, 32); ss += __shfl_xor(ss, m, 32); }
  float mean = s * (1.f/512.f);
  float var  = ss * (1.f/512.f) - mean*mean;
  float rstd = rsqrtf(var + 1e-5f);
  OT* o = out + (size_t)row*outstride;
#pragma unroll
  for (int i = 0; i < 16; i++) { int c = lane + i*32; o[c] = (OT)((p[c]-mean)*rstd*gam[c] + bet[c]); }
}

// ---------------------------------------------------------------------------
// Patch gather: video (4,8,3,224,224) f32 -> vp (6272, 768) f16
// ---------------------------------------------------------------------------
__global__ void k_patch_gather(const float* __restrict__ video, _Float16* __restrict__ vp)
{
  int idx = blockIdx.x*256 + threadIdx.x;
  const int total = NB*NTOK*768;
  if (idx >= total) return;
  int col = idx % 768;
  int row = idx / 768;
  int bb = row / NTOK, rem = row % NTOK;
  int f  = rem / NPATCH, pr = rem % NPATCH;
  int hpi = pr / 14, wpi = pr % 14;
  int cc = col % 3, pix = col / 3;
  int px = pix % 16, py = pix / 16;
  size_t src = ((((size_t)(bb*NF + f))*3 + cc)*224 + (hpi*16 + py))*224 + (wpi*16 + px);
  vp[idx] = (_Float16)video[src];
}

// ---------------------------------------------------------------------------
// Assemble x = [cls, aa_w@aa_W+aa_b, tokens] : (4, 1570, 512) f32
// ---------------------------------------------------------------------------
__global__ void k_assemble(const float* __restrict__ tokens, const float* __restrict__ cls,
    const float* __restrict__ aa_w, const float* __restrict__ aa_W,
    const float* __restrict__ aa_b, float* __restrict__ x)
{
  int idx = blockIdx.x*256 + threadIdx.x;
  const int total = NB*SEQ*DMODEL;
  if (idx >= total) return;
  int d = idx % DMODEL;
  int rr = idx / DMODEL;
  int s = rr % SEQ, bb = rr / SEQ;
  float v;
  if (s == 0) v = cls[d];
  else if (s == 1) {
    v = aa_b[d];
    for (int j = 0; j < 18; j++) v += aa_w[bb*18 + j] * aa_W[j*DMODEL + d];
  } else {
    v = tokens[((size_t)bb*NTOK + (s-2))*DMODEL + d];
  }
  x[idx] = v;
}

// ---------------------------------------------------------------------------
// Rotary tables
// ---------------------------------------------------------------------------
__global__ void k_frot(float* __restrict__ sn, float* __restrict__ cs)
{
  int idx = blockIdx.x*256 + threadIdx.x;
  if (idx >= NF*DHEAD) return;
  int t = idx / DHEAD, j = idx % DHEAD, i = j >> 1;
  float inv = powf(10000.f, -((float)(2*i) / 64.f));
  float fr = (float)t * inv;
  sn[idx] = sinf(fr); cs[idx] = cosf(fr);
}

__global__ void k_irot(float* __restrict__ sn, float* __restrict__ cs)
{
  int idx = blockIdx.x*256 + threadIdx.x;
  if (idx >= NPATCH*DHEAD) return;
  int p = idx / DHEAD, j = idx % DHEAD, jh = j >> 1;
  int h = p / 14, w = p % 14;
  int sidx = (jh < 16) ? jh : (jh - 16);
  float scale = 1.f + 4.f * ((float)sidx / 15.f);          // linspace(1,5,16)
  float coord = (jh < 16) ? (-1.f + 2.f*(float)h/13.f) : (-1.f + 2.f*(float)w/13.f);
  float v = coord * scale * 3.14159265358979f;
  sn[idx] = sinf(v); cs[idx] = cosf(v);
}

// ---------------------------------------------------------------------------
// QKV -> per-head Q (scaled, rotated), K (rotated), K/V raw. mode 0=time 1=space
// ---------------------------------------------------------------------------
__global__ void k_split_heads(const float* __restrict__ qkv,
    const float* __restrict__ sinT, const float* __restrict__ cosT,
    float* __restrict__ Qr, float* __restrict__ Kr,
    float* __restrict__ Kn, float* __restrict__ Vn, int mode)
{
  int idx = blockIdx.x*256 + threadIdx.x;       // pair index
  const int total = NBH*SEQ*(DHEAD/2);
  if (idx >= total) return;
  int i = idx & 31;
  int rem = idx >> 5;
  int s  = rem % SEQ;
  int bh = rem / SEQ;
  int bb = bh >> 3, h = bh & 7;
  size_t src = ((size_t)(bb*SEQ + s))*(3*DMODEL) + h*DHEAD + 2*i;
  float q0 = qkv[src]        * 0.125f, q1 = qkv[src+1]      * 0.125f;   // DH^-0.5
  float k0 = qkv[src+DMODEL],          k1 = qkv[src+DMODEL+1];
  float v0 = qkv[src+2*DMODEL],        v1 = qkv[src+2*DMODEL+1];
  size_t dst = ((size_t)bh*SEQ + s)*DHEAD + 2*i;
  Kn[dst] = k0; Kn[dst+1] = k1;
  Vn[dst] = v0; Vn[dst+1] = v1;
  float qr0 = q0, qr1 = q1, kr0 = k0, kr1 = k1;
  if (s >= 2) {
    int t = s - 2;
    int pos = (mode == 0) ? (t / NPATCH) : (t % NPATCH);
    const float* sp = sinT + pos*DHEAD + 2*i;
    const float* cp = cosT + pos*DHEAD + 2*i;
    float s0 = sp[0], s1 = sp[1], c0 = cp[0], c1 = cp[1];
    qr0 = q0*c0 - q1*s0;  qr1 = q1*c1 + q0*s1;   // t*cos + rotate_every_two(t)*sin
    kr0 = k0*c0 - k1*s0;  kr1 = k1*c1 + k0*s1;
  }
  Qr[dst] = qr0; Qr[dst+1] = qr1;
  Kr[dst] = kr0; Kr[dst+1] = kr1;
}

// ---------------------------------------------------------------------------
// Softmax attention, one block per query. mode 0=time(Lk=10) 1=space(Lk=198)
// 2=special(Lk=1570, keys are raw Kn). grid=(groups, Lq).
// ---------------------------------------------------------------------------
__global__ void k_attention(const float* __restrict__ Q, const float* __restrict__ Km,
    const float* __restrict__ V, float* __restrict__ O,
    int mode, int auxdiv, int Lk)
{
  __shared__ float sc[1600];
  __shared__ float rbuf[256];
  __shared__ float qs[64];
  const int tid  = threadIdx.x;
  const int nthr = blockDim.x;
  const int grp  = blockIdx.x;
  const int qi   = blockIdx.y;
  const int bh   = grp / auxdiv;
  const int aux  = grp % auxdiv;
  int sq;
  if (mode == 0)      sq = 2 + qi*NPATCH + aux;
  else if (mode == 1) sq = 2 + aux*NPATCH + qi;
  else                sq = qi;
  const size_t base = (size_t)bh * SEQ * DHEAD;
  const float* qp = Q + base + (size_t)sq*DHEAD;
  if (tid < 64) qs[tid] = qp[tid];
  __syncthreads();

  float lmax = -3.0e38f;
  for (int j = tid; j < Lk; j += nthr) {
    int kr;
    if (mode == 2 || j < 2) kr = j;
    else if (mode == 0)     kr = 2 + (j-2)*NPATCH + aux;
    else                    kr = 2 + aux*NPATCH + (j-2);
    const float* kp = Km + base + (size_t)kr*DHEAD;
    float d = 0.f;
#pragma unroll 8
    for (int t = 0; t < DHEAD; t++) d += qs[t]*kp[t];
    sc[j] = d;
    lmax = fmaxf(lmax, d);
  }
  rbuf[tid] = lmax; __syncthreads();
  for (int off = nthr>>1; off > 0; off >>= 1) {
    if (tid < off) rbuf[tid] = fmaxf(rbuf[tid], rbuf[tid+off]);
    __syncthreads();
  }
  float mx = rbuf[0];
  __syncthreads();
  float lsum = 0.f;
  for (int j = tid; j < Lk; j += nthr) {
    float e = __expf(sc[j] - mx);
    sc[j] = e; lsum += e;
  }
  rbuf[tid] = lsum; __syncthreads();
  for (int off = nthr>>1; off > 0; off >>= 1) {
    if (tid < off) rbuf[tid] += rbuf[tid+off];
    __syncthreads();
  }
  float inv = 1.0f / rbuf[0];
  __syncthreads();
  const int d  = tid & 63;
  const int kg = tid >> 6;
  const int nkg = nthr >> 6;
  float a0 = 0.f;
  for (int j = kg; j < Lk; j += nkg) {
    int kr;
    if (mode == 2 || j < 2) kr = j;
    else if (mode == 0)     kr = 2 + (j-2)*NPATCH + aux;
    else                    kr = 2 + aux*NPATCH + (j-2);
    a0 += sc[j] * V[base + (size_t)kr*DHEAD + d];
  }
  rbuf[tid] = a0; __syncthreads();
  if (tid < 64) {
    float s = 0.f;
    for (int k = 0; k < nkg; k++) s += rbuf[k*64 + tid];
    O[base + (size_t)sq*DHEAD + tid] = s * inv;
  }
}

// ---------------------------------------------------------------------------
// Head merge: O (32,1570,64) f32 -> M (6280,512) f16 (GEMM A operand)
// ---------------------------------------------------------------------------
__global__ void k_merge(const float* __restrict__ O, _Float16* __restrict__ Mo)
{
  int idx = blockIdx.x*256 + threadIdx.x;
  const int total = NROWS*DMODEL;
  if (idx >= total) return;
  int col = idx % DMODEL;
  int row = idx / DMODEL;
  int s = row % SEQ, bb = row / SEQ;
  int h = col / DHEAD, d = col % DHEAD;
  Mo[idx] = (_Float16)O[(((size_t)(bb*NH + h))*SEQ + s)*DHEAD + d];
}

// ---------------------------------------------------------------------------
// GEGLU: hdn (6280,4096) f32 -> h2 (6280,2048) f16, exact erf gelu on gate
// ---------------------------------------------------------------------------
__global__ void k_geglu(const float* __restrict__ hdn, _Float16* __restrict__ h2)
{
  int idx = blockIdx.x*256 + threadIdx.x;
  const int total = NROWS*2048;
  if (idx >= total) return;
  int c = idx % 2048;
  size_t row = (size_t)(idx / 2048);
  float a = hdn[row*4096 + c];
  float g = hdn[row*4096 + 2048 + c];
  float ge = 0.5f * g * (1.f + erff(g * 0.70710678118654752f));
  h2[idx] = (_Float16)(a * ge);
}

// ---------------------------------------------------------------------------
// Final head: (4,512) @ (512,2) + b -> d_out, one wave per output element
// ---------------------------------------------------------------------------
__global__ __launch_bounds__(256) void k_head_out(const float* __restrict__ x0,
    const float* __restrict__ W, const float* __restrict__ bvec, float* __restrict__ out)
{
  int wv = threadIdx.x >> 5, lane = threadIdx.x & 31;
  int bb = wv >> 1, o = wv & 1;
  const float* p = x0 + bb*DMODEL;
  float s = 0.f;
  for (int i = lane; i < DMODEL; i += 32) s += p[i]*W[i*2 + o];
#pragma unroll
  for (int m = 16; m > 0; m >>= 1) s += __shfl_xor(s, m, 32);
  if (lane == 0) out[bb*2 + o] = s + bvec[o];
}

// ---------------------------------------------------------------------------
// Host orchestration
// ---------------------------------------------------------------------------
extern "C" void kernel_launch(void* const* d_in, const int* in_sizes, int n_in,
                              void* d_out, int out_size, void* d_ws, size_t ws_size,
                              hipStream_t stream)
{
  (void)in_sizes; (void)n_in; (void)out_size; (void)ws_size;
  const float* video   = (const float*)d_in[0];
  const float* aa_w    = (const float*)d_in[1];
  const float* patch_W = (const float*)d_in[2];
  const float* patch_b = (const float*)d_in[3];
  const float* cls_tok = (const float*)d_in[4];
  const float* aa_W    = (const float*)d_in[5];
  const float* aa_b    = (const float*)d_in[6];
  const float* ln_t_g  = (const float*)d_in[7];
  const float* ln_t_b  = (const float*)d_in[8];
  const float* t_qkv   = (const float*)d_in[9];
  const float* t_outW  = (const float*)d_in[10];
  const float* t_outb  = (const float*)d_in[11];
  const float* ln_s_g  = (const float*)d_in[12];
  const float* ln_s_b  = (const float*)d_in[13];
  const float* s_qkv   = (const float*)d_in[14];
  const float* s_outW  = (const float*)d_in[15];
  const float* s_outb  = (const float*)d_in[16];
  const float* ln_f_g  = (const float*)d_in[17];
  const float* ln_f_b  = (const float*)d_in[18];
  const float* ff_W1   = (const float*)d_in[19];
  const float* ff_b1   = (const float*)d_in[20];
  const float* ff_W2   = (const float*)d_in[21];
  const float* ff_b2   = (const float*)d_in[22];
  const float* out_ln_g= (const float*)d_in[23];
  const float* out_ln_b= (const float*)d_in[24];
  const float* out_W   = (const float*)d_in[25];
  const float* out_b   = (const float*)d_in[26];

  float* ws = (float*)d_ws;
  size_t off = 0;
  auto falloc = [&](size_t n) { float* p = ws + off; off += n; return p; };
  auto halloc = [&](size_t n) { _Float16* p = (_Float16*)(ws + off); off += (n + 1)/2; return p; };
  const size_t XSZ = (size_t)NROWS*DMODEL;          // 3,215,360
  float*    x      = falloc(XSZ);
  float*    qkv    = falloc((size_t)NROWS*3*DMODEL);
  float*    Qr     = falloc(XSZ);
  float*    Kr     = falloc(XSZ);
  float*    Kn     = falloc(XSZ);
  float*    Vn     = falloc(XSZ);
  float*    Ohead  = falloc(XSZ);
  float*    hdn    = falloc((size_t)NROWS*4096);
  float*    tokens = falloc((size_t)NB*NTOK*DMODEL);
  _Float16* xln_h  = halloc(XSZ);
  _Float16* mrg_h  = halloc(XSZ);
  _Float16* h2_h   = halloc((size_t)NROWS*2048);
  _Float16* vp_h   = halloc((size_t)NB*NTOK*768);
  _Float16* Bh     = halloc((size_t)DMODEL*4096);   // largest weight (f16)
  float*    fsin   = falloc(NF*DHEAD);
  float*    fcos   = falloc(NF*DHEAD);
  float*    isin   = falloc(NPATCH*DHEAD);
  float*    icos   = falloc(NPATCH*DHEAD);
  float*    x0ln   = falloc(NB*DMODEL);

  auto gemm = [&](const _Float16* A, const _Float16* Bm, const float* bias,
                  const float* resid, float* C, int M, int N, int K, int flags) {
    dim3 grid((N + BN - 1)/BN, (M + BM - 1)/BM);
    k_gemm<<<grid, dim3(256), 0, stream>>>(A, Bm, bias, resid, C, M, N, K, flags);
  };
  auto castW = [&](const float* src, int n) {       // n % 4 == 0 for all weights
    int n4 = n / 4;
    k_cast_f16<<<(n4 + 255)/256, 256, 0, stream>>>(src, Bh, n4);
  };
  auto lnorm_h = [&](const float* in, const float* g, const float* b) {
    k_layernorm<_Float16><<<(NROWS + 7)/8, 256, 0, stream>>>(in, DMODEL, g, b, xln_h, DMODEL, NROWS);
  };

  // ---- stem ----
  k_patch_gather<<<(NB*NTOK*768 + 255)/256, 256, 0, stream>>>(video, vp_h);
  castW(patch_W, 768*DMODEL);
  gemm(vp_h, Bh, patch_b, nullptr, tokens, NB*NTOK, DMODEL, 768, 1);
  k_assemble<<<(NB*SEQ*DMODEL + 255)/256, 256, 0, stream>>>(tokens, cls_tok, aa_w, aa_W, aa_b, x);
  k_frot<<<2, 256, 0, stream>>>(fsin, fcos);
  k_irot<<<(NPATCH*DHEAD + 255)/256, 256, 0, stream>>>(isin, icos);

  const int PAIRS = NBH*SEQ*(DHEAD/2);

  auto attention = [&](const float* lng, const float* lnb, const float* Wqkv,
                       const float* Wout, const float* bout, int mode,
                       const float* sn, const float* cn) {
    lnorm_h(x, lng, lnb);
    castW(Wqkv, DMODEL*3*DMODEL);
    gemm(xln_h, Bh, nullptr, nullptr, qkv, NROWS, 3*DMODEL, DMODEL, 0);
    k_split_heads<<<(PAIRS + 255)/256, 256, 0, stream>>>(qkv, sn, cn, Qr, Kr, Kn, Vn, mode);
    // special tokens attend over full, unrotated K/V
    k_attention<<<dim3(NBH, 2), 256, 0, stream>>>(Qr, Kn, Vn, Ohead, 2, 1, SEQ);
    if (mode == 0)
      k_attention<<<dim3(NBH*NPATCH, NF), 128, 0, stream>>>(Qr, Kr, Vn, Ohead, 0, NPATCH, 2 + NF);
    else
      k_attention<<<dim3(NBH*NF, NPATCH), 128, 0, stream>>>(Qr, Kr, Vn, Ohead, 1, NF, 2 + NPATCH);
    k_merge<<<(NROWS*DMODEL + 255)/256, 256, 0, stream>>>(Ohead, mrg_h);
    castW(Wout, DMODEL*DMODEL);
    gemm(mrg_h, Bh, bout, x, x, NROWS, DMODEL, DMODEL, 3);  // bias + residual
  };

  for (int L = 0; L < 12; L++) {
    attention(ln_t_g + (size_t)L*DMODEL, ln_t_b + (size_t)L*DMODEL,
              t_qkv + (size_t)L*DMODEL*3*DMODEL,
              t_outW + (size_t)L*DMODEL*DMODEL, t_outb + (size_t)L*DMODEL,
              0, fsin, fcos);
    attention(ln_s_g + (size_t)L*DMODEL, ln_s_b + (size_t)L*DMODEL,
              s_qkv + (size_t)L*DMODEL*3*DMODEL,
              s_outW + (size_t)L*DMODEL*DMODEL, s_outb + (size_t)L*DMODEL,
              1, isin, icos);
    // feed-forward (GEGLU)
    lnorm_h(x, ln_f_g + (size_t)L*DMODEL, ln_f_b + (size_t)L*DMODEL);
    castW(ff_W1 + (size_t)L*DMODEL*4096, DMODEL*4096);
    gemm(xln_h, Bh, ff_b1 + (size_t)L*4096, nullptr, hdn, NROWS, 4096, DMODEL, 1);
    k_geglu<<<(NROWS*2048 + 255)/256, 256, 0, stream>>>(hdn, h2_h);
    castW(ff_W2 + (size_t)L*2048*DMODEL, 2048*DMODEL);
    gemm(h2_h, Bh, ff_b2 + (size_t)L*DMODEL, x, x, NROWS, DMODEL, 2048, 3);  // bias + residual
  }

  // ---- head ----
  k_layernorm<float><<<1, 256, 0, stream>>>(x, (long long)SEQ*DMODEL,
                                            out_ln_g, out_ln_b, x0ln, DMODEL, NB);
  k_head_out<<<1, 256, 0, stream>>>(x0ln, out_W, out_b, (float*)d_out);
}
